// GCNNet_57621281243147
// MI455X (gfx1250) — matrix operand order repaired and checked by reading, not compiled, not run on hardware
//
#include <hip/hip_runtime.h>
#include <cstdint>

#define FEAT    133
#define KPAD    160          // K padded to 5 tiles of 32
#define NT      9            // 9 N-tiles of 16 (144 >= 133)
#define KT      5            // 5 K-tiles of 32
#define NGRAPH  256
#define WFRAG   (KT * NT * 32)       // 1440 v16bf fragments per (hi|lo) half
#define WELEMS  (WFRAG * 16)         // 23040 bf16 elements per half

typedef __attribute__((ext_vector_type(16))) __bf16 v16bf;
typedef __attribute__((ext_vector_type(8)))  __bf16 v8bf;
typedef __attribute__((ext_vector_type(8)))  float  v8f;

// fp32 -> (bf16 hi, bf16 lo) truncation split; hi+lo carries ~16 mantissa bits.
static __device__ __forceinline__ void split16(float f, unsigned short& h, unsigned short& l) {
    unsigned u  = __builtin_bit_cast(unsigned, f);
    unsigned hu = u & 0xffff0000u;
    h = (unsigned short)(hu >> 16);
    float rem = f - __builtin_bit_cast(float, hu);
    l = (unsigned short)(__builtin_bit_cast(unsigned, rem) >> 16);
}

// ---------------- degree / dinv ----------------
__global__ void k_fill1(float* p, int n) {
    int i = blockIdx.x * blockDim.x + threadIdx.x;
    if (i < n) p[i] = 1.0f;                 // self-loop contributes 1
}

__global__ void k_degcount(const long long* __restrict__ ei, int E, float* deg) {
    int e = blockIdx.x * blockDim.x + threadIdx.x;
    if (e < E) unsafeAtomicAdd(&deg[ei[E + e]], 1.0f);
}

__global__ void k_rsqrt(float* p, int n) {
    int i = blockIdx.x * blockDim.x + threadIdx.x;
    if (i < n) p[i] = __frsqrt_rn(p[i]);
}

// ---------------- prep: pad + bf16-split activations ----------------
__global__ void k_prepX(const float* __restrict__ X, unsigned short* __restrict__ xhi,
                        unsigned short* __restrict__ xlo, int N) {
    int idx = blockIdx.x * blockDim.x + threadIdx.x;
    if (idx >= N * KPAD) return;
    int row = idx / KPAD;
    int k   = idx - row * KPAD;
    float f = (k < FEAT) ? X[(size_t)row * FEAT + k] : 0.0f;
    unsigned short h, l; split16(f, h, l);
    xhi[idx] = h; xlo[idx] = l;
}

// ---------------- prep: W -> wave32 B-fragment-swizzled bf16 hi/lo ----------------
// wsw[((kt*NT+nt)*32+lane)*16 + i] ; hi half then lo half (WELEMS each)
__global__ void k_prepW(const float* __restrict__ W, unsigned short* __restrict__ wsw) {
    int tid = blockIdx.x * blockDim.x + threadIdx.x;
    if (tid >= WELEMS) return;
    int i    = tid & 15;
    int lane = (tid >> 4) & 31;
    int nt   = (tid >> 9) % NT;
    int kt5  = (tid >> 9) / NT;
    int k = kt5 * 32 + ((lane >> 4) << 3) + (i & 7) + ((i >> 3) << 4);   // lanes 16-31: K+8 block, elems 8-15: K+16
    int n = (nt << 4) + (lane & 15);
    float f = (k < FEAT && n < FEAT) ? W[k * FEAT + n] : 0.0f;
    unsigned short h, l; split16(f, h, l);
    wsw[tid] = h;
    wsw[WELEMS + tid] = l;
}

// ---------------- GEMM: g = (X @ W) * dinv[row] ----------------
// Wave per 16-row M tile, 9 N-tile register block, B fragments staged in LDS.
__global__ void __launch_bounds__(128)
k_gemm_g(const unsigned short* __restrict__ xhi, const unsigned short* __restrict__ xlo,
         const unsigned short* __restrict__ wsw,
         const float* __restrict__ dinv, float* __restrict__ g, int nrows) {
    __shared__ v16bf sW[2 * WFRAG];              // 92160 B of LDS

    // linear cooperative copy of swizzled W (hi+lo) into LDS
    {
        const uint4* src = (const uint4*)wsw;
        uint4*       dst = (uint4*)sW;
        const int n16 = (2 * WELEMS * 2) / 16;   // 5760 x 16B
        for (int t = threadIdx.x; t < n16; t += blockDim.x) dst[t] = src[t];
    }
    __syncthreads();

    const int mtiles = (nrows + 15) >> 4;
    const int wave   = blockIdx.x * (blockDim.x >> 5) + (threadIdx.x >> 5);
    if (wave < mtiles) {
        const int lane = threadIdx.x & 31;
        const int l15  = lane & 15;
        const int hi16 = lane >> 4;
        const int m0   = wave << 4;

        v8f acc[NT];
#pragma unroll
        for (int t = 0; t < NT; ++t)
#pragma unroll
            for (int r = 0; r < 8; ++r) acc[t][r] = 0.0f;

        int arow = m0 + l15; if (arow >= nrows) arow = nrows - 1;
        const unsigned short* xh = xhi + (size_t)arow * KPAD;
        const unsigned short* xl = xlo + (size_t)arow * KPAD;

#pragma unroll
        for (int kt5 = 0; kt5 < KT; ++kt5) {
            const int ka0 = kt5 * 32 + (hi16 << 3);   // lanes 16-31: K+8 block
            v8bf a0h = *(const v8bf*)(xh + ka0);
            v8bf a1h = *(const v8bf*)(xh + ka0 + 16);
            v8bf a0l = *(const v8bf*)(xl + ka0);
            v8bf a1l = *(const v8bf*)(xl + ka0 + 16);
            v16bf ahi = __builtin_shufflevector(a0h, a1h, 0,1,2,3,4,5,6,7,8,9,10,11,12,13,14,15);
            v16bf alo = __builtin_shufflevector(a0l, a1l, 0,1,2,3,4,5,6,7,8,9,10,11,12,13,14,15);
#pragma unroll
            for (int nt = 0; nt < NT; ++nt) {
                const int fi = (kt5 * NT + nt) * 32 + lane;
                v16bf bhi = sW[fi];
                v16bf blo = sW[WFRAG + fi];
                acc[nt] = __builtin_amdgcn_wmma_f32_16x16x32_bf16(
                    false, ahi, false, bhi, (short)0, acc[nt], false, false);
                acc[nt] = __builtin_amdgcn_wmma_f32_16x16x32_bf16(
                    false, alo, false, bhi, (short)0, acc[nt], false, false);
                acc[nt] = __builtin_amdgcn_wmma_f32_16x16x32_bf16(
                    false, ahi, false, blo, (short)0, acc[nt], false, false);
            }
        }

        // epilogue: scale by dinv[row], masked store (col < 133)
#pragma unroll
        for (int nt = 0; nt < NT; ++nt) {
            const int col = (nt << 4) + l15;
#pragma unroll
            for (int r = 0; r < 8; ++r) {
                const int rr = m0 + r + (hi16 << 3);  // C layout: vgpr r -> M=r / M=8+r
                if (col < FEAT && rr < nrows)
                    g[(size_t)rr * FEAT + col] = acc[nt][r] * dinv[rr];
            }
        }
    }
}

// ---------------- buffer copy (agg init = self-loop message) ----------------
__global__ void k_copy4(const float4* __restrict__ s, float4* __restrict__ d, int n4) {
    int i = blockIdx.x * blockDim.x + threadIdx.x;
    if (i < n4) d[i] = s[i];
}

// ---------------- edge scatter: agg[dst] += g[src] ----------------
__global__ void k_scatter(const long long* __restrict__ ei, int E,
                          const float* __restrict__ g, float* __restrict__ agg) {
    int e = blockIdx.x * (blockDim.x >> 5) + (threadIdx.x >> 5);
    if (e >= E) return;
    const int lane = threadIdx.x & 31;
    const long long s = ei[e];
    const long long d = ei[E + e];
    const float* gs = g + (size_t)s * FEAT;
    float* ad = agg + (size_t)d * FEAT;
#pragma unroll
    for (int f = lane; f < FEAT; f += 32)
        unsafeAtomicAdd(&ad[f], gs[f]);
}

// ---------------- epilogues: relu(agg*dinv + b) ----------------
// layer-1 variant writes straight into the padded bf16 split buffers
__global__ void k_epi_bf(const float* __restrict__ agg, const float* __restrict__ dinv,
                         const float* __restrict__ b,
                         unsigned short* __restrict__ xhi, unsigned short* __restrict__ xlo,
                         int nf) {
    int idx = blockIdx.x * blockDim.x + threadIdx.x;
    if (idx >= nf) return;
    int i = idx / FEAT;
    int f = idx - i * FEAT;
    float v = fmaxf(agg[idx] * dinv[i] + b[f], 0.0f);
    unsigned short h, l; split16(v, h, l);
    xhi[(size_t)i * KPAD + f] = h;
    xlo[(size_t)i * KPAD + f] = l;
}

__global__ void k_epi_f32(const float* __restrict__ agg, const float* __restrict__ dinv,
                          const float* __restrict__ b, float* __restrict__ out, int nf) {
    int idx = blockIdx.x * blockDim.x + threadIdx.x;
    if (idx >= nf) return;
    int i = idx / FEAT;
    int f = idx - i * FEAT;
    out[idx] = fmaxf(agg[idx] * dinv[i] + b[f], 0.0f);
}

// ---------------- pooling ----------------
__global__ void k_zero(float* p, int n) {
    int i = blockIdx.x * blockDim.x + threadIdx.x;
    if (i < n) p[i] = 0.0f;
}

__global__ void k_poolacc(const float* __restrict__ x, const long long* __restrict__ batch,
                          int N, float* __restrict__ pool, float* __restrict__ cnt) {
    int i = blockIdx.x * (blockDim.x >> 5) + (threadIdx.x >> 5);
    if (i >= N) return;
    const int lane = threadIdx.x & 31;
    const long long bg = batch[i];
    const float* xr = x + (size_t)i * FEAT;
    float* pr = pool + (size_t)bg * FEAT;
#pragma unroll
    for (int f = lane; f < FEAT; f += 32)
        unsafeAtomicAdd(&pr[f], xr[f]);
    if (lane == 0) unsafeAtomicAdd(&cnt[bg], 1.0f);
}

__global__ void k_poolfin(const float* __restrict__ pool, const float* __restrict__ cnt,
                          float* __restrict__ out, int n) {
    int idx = blockIdx.x * blockDim.x + threadIdx.x;
    if (idx >= n) return;
    int g = idx / FEAT;
    out[idx] = pool[idx] / fmaxf(cnt[g], 1.0f);
}

static inline int cdiv(long long a, long long b) { return (int)((a + b - 1) / b); }

extern "C" void kernel_launch(void* const* d_in, const int* in_sizes, int n_in,
                              void* d_out, int out_size, void* d_ws, size_t ws_size,
                              hipStream_t stream) {
    const float*     x     = (const float*)d_in[0];
    const long long* ei    = (const long long*)d_in[1];
    const long long* batch = (const long long*)d_in[2];
    const float*     W1    = (const float*)d_in[3];
    const float*     b1    = (const float*)d_in[4];
    const float*     W2    = (const float*)d_in[5];
    const float*     b2    = (const float*)d_in[6];
    float*           out   = (float*)d_out;

    const int N  = in_sizes[2];
    const int E  = in_sizes[1] / 2;
    const size_t NF = (size_t)N * FEAT;

    float* dinv = (float*)d_ws;                       // N
    float* bufA = dinv + N;                           // N*FEAT
    float* bufB = bufA + NF;                          // N*FEAT
    float* pool = bufB + NF;                          // NGRAPH*FEAT
    float* cnt  = pool + NGRAPH * FEAT;               // NGRAPH
    unsigned short* xhi  = (unsigned short*)(cnt + NGRAPH);   // N*KPAD
    unsigned short* xlo  = xhi + (size_t)N * KPAD;            // N*KPAD
    unsigned short* wsw1 = xlo + (size_t)N * KPAD;            // 2*WELEMS
    unsigned short* wsw2 = wsw1 + 2 * WELEMS;                 // 2*WELEMS

    const int mtiles = (N + 15) / 16;

    // degrees -> dinv (shared by both layers)
    k_fill1   <<<cdiv(N, 256), 256, 0, stream>>>(dinv, N);
    k_degcount<<<cdiv(E, 256), 256, 0, stream>>>(ei, E, dinv);
    k_rsqrt   <<<cdiv(N, 256), 256, 0, stream>>>(dinv, N);

    // one-time preps
    k_prepX<<<cdiv((long long)N * KPAD, 256), 256, 0, stream>>>(x, xhi, xlo, N);
    k_prepW<<<cdiv(WELEMS, 256), 256, 0, stream>>>(W1, wsw1);
    k_prepW<<<cdiv(WELEMS, 256), 256, 0, stream>>>(W2, wsw2);

    // ---- layer 1 ----
    k_gemm_g <<<cdiv(mtiles, 4), 128, 0, stream>>>(xhi, xlo, wsw1, dinv, bufA, N);
    k_copy4  <<<cdiv((long long)(NF / 4), 256), 256, 0, stream>>>(
                 (const float4*)bufA, (float4*)bufB, (int)(NF / 4));
    k_scatter<<<cdiv(E, 8), 256, 0, stream>>>(ei, E, bufA, bufB);
    k_epi_bf <<<cdiv((long long)NF, 256), 256, 0, stream>>>(bufB, dinv, b1, xhi, xlo, (int)NF);

    // ---- layer 2 ----
    k_gemm_g <<<cdiv(mtiles, 4), 128, 0, stream>>>(xhi, xlo, wsw2, dinv, bufB, N);
    k_copy4  <<<cdiv((long long)(NF / 4), 256), 256, 0, stream>>>(
                 (const float4*)bufB, (float4*)bufA, (int)(NF / 4));
    k_scatter<<<cdiv(E, 8), 256, 0, stream>>>(ei, E, bufB, bufA);
    k_epi_f32<<<cdiv((long long)NF, 256), 256, 0, stream>>>(bufA, dinv, b2, bufB, (int)NF);

    // ---- global mean pool ----
    k_zero   <<<cdiv(NGRAPH * FEAT + NGRAPH, 256), 256, 0, stream>>>(pool, NGRAPH * FEAT + NGRAPH);
    k_poolacc<<<cdiv(N, 8), 256, 0, stream>>>(bufB, batch, N, pool, cnt);
    k_poolfin<<<cdiv(NGRAPH * FEAT, 256), 256, 0, stream>>>(pool, cnt, out, NGRAPH * FEAT);
}
